// TransformerEncoderLayer_89489938579620
// MI455X (gfx1250) — matrix operand order repaired and checked by reading, not compiled, not run on hardware
//
#include <hip/hip_runtime.h>
#include <math.h>

// ---------------------------------------------------------------------------
// Types / helpers
// ---------------------------------------------------------------------------
typedef unsigned short u16t;
typedef __attribute__((ext_vector_type(16))) __bf16 v16bf;
typedef __attribute__((ext_vector_type(8)))  float  v8f;
typedef __attribute__((ext_vector_type(4)))  unsigned int v4u;
typedef __attribute__((ext_vector_type(8)))  unsigned int v8u;

union FragB {
  v16bf v;
  uint4 u[2];
};

__device__ __forceinline__ u16t f2bf(float f) {
  unsigned int u = __float_as_uint(f);
  u += 0x7fffu + ((u >> 16) & 1u);  // round-to-nearest-even
  return (u16t)(u >> 16);
}

__device__ __forceinline__ v8f wmma_bf16(FragB a, FragB b, v8f c) {
  return __builtin_amdgcn_wmma_f32_16x16x32_bf16(
      false, a.v, false, b.v, (short)0, c, false, false);
}

// gfx1250 async memory->LDS copy (per-lane; tracked by ASYNCcnt)
__device__ __forceinline__ void async_b128(u16t* lds, const u16t* g) {
  unsigned ldsa = (unsigned)(unsigned long long)lds;  // low 32 bits = LDS offset
  asm volatile("global_load_async_to_lds_b128 %0, %1, off"
               :: "v"(ldsa), "v"(g) : "memory");
}
__device__ __forceinline__ void wait_async() {
  asm volatile("s_wait_asynccnt 0x0" ::: "memory");
}

// gfx1250 Tensor Data Mover: one instruction DMAs a 2D tile (tile1 rows x
// tile0 elements, row stride stride0 elements) of bf16 into LDS, row-major
// contiguous.  D# per CDNA5 ISA ch.8 (groups 2/3 = NULL -> 2D tensor).
__device__ __forceinline__ void tdm_load_2d(u16t* lds, const u16t* g,
                                            unsigned tile0, unsigned tile1,
                                            unsigned long long stride0) {
  unsigned ldsa = (unsigned)(unsigned long long)lds;
  unsigned long long ga = (unsigned long long)g;
  v4u g0;
  g0[0] = 1u;                                        // count=1 (valid), user mode
  g0[1] = ldsa;                                      // lds_addr [63:32]
  g0[2] = (unsigned)(ga & 0xffffffffu);              // global_addr [95:64]
  g0[3] = (unsigned)((ga >> 32) & 0x01ffffffu)       // global_addr [120:96]
          | 0x80000000u;                             // type=2 ("image") [127:126]
  v8u g1;
  g1[0] = 0x10000u;                                  // wg_mask=0, data_size=1 (2B)
  g1[1] = (tile0 & 0xffffu) << 16;                   // tensor_dim0[15:0] @ [63:48]
  g1[2] = (tile0 >> 16) | ((tile1 & 0xffffu) << 16); // tdim0[31:16] | tdim1[15:0]
  g1[3] = (tile1 >> 16) | (tile0 << 16);             // tdim1[31:16] | tile_dim0
  g1[4] = tile1 & 0xffffu;                           // tile_dim1 | tile_dim2=0
  g1[5] = (unsigned)(stride0 & 0xffffffffu);         // tensor_dim0_stride[31:0]
  g1[6] = (unsigned)((stride0 >> 32) & 0xffffu);     // stride0[47:32] | stride1.lo=0
  g1[7] = 0u;                                        // stride1.hi
  asm volatile("tensor_load_to_lds %0, %1" :: "s"(g0), "s"(g1) : "memory");
}
__device__ __forceinline__ void wait_tensor() {
  asm volatile("s_wait_tensorcnt 0x0" ::: "memory");
}

#define Bn   8
#define Tn   1024
#define Dm   1024
#define Hn   16
#define Dff  4096
#define Rk   64
#define BT   (Bn * Tn)
#define LSCALE (1.0f / 64.0f)

// ---------------------------------------------------------------------------
// f32 -> bf16 conversion
// ---------------------------------------------------------------------------
__global__ void cvt_kernel(const float* __restrict__ in, u16t* __restrict__ out, long n) {
  long i  = (long)blockIdx.x * blockDim.x + threadIdx.x;
  long st = (long)gridDim.x * blockDim.x;
  for (; i < n; i += st) out[i] = f2bf(in[i]);
}

__global__ void gelu_cvt_kernel(const float* __restrict__ in, u16t* __restrict__ out, long n) {
  long i  = (long)blockIdx.x * blockDim.x + threadIdx.x;
  long st = (long)gridDim.x * blockDim.x;
  for (; i < n; i += st) {
    float v = in[i];
    out[i] = f2bf(0.5f * v * (1.0f + erff(v * 0.70710678118654752f)));
  }
}

// ---------------------------------------------------------------------------
// LayerNorm -> bf16.  One block per row.
// ---------------------------------------------------------------------------
__global__ void __launch_bounds__(256)
ln_kernel(const float* __restrict__ x, const float* __restrict__ g, u16t* __restrict__ out) {
  const long row = blockIdx.x;
  const float* xr = x + row * Dm;
  const int tid = threadIdx.x;
  float4 v = *reinterpret_cast<const float4*>(xr + tid * 4);
  float s  = v.x + v.y + v.z + v.w;
  float ss = v.x * v.x + v.y * v.y + v.z * v.z + v.w * v.w;
#pragma unroll
  for (int m = 1; m < 32; m <<= 1) { s += __shfl_xor(s, m); ss += __shfl_xor(ss, m); }
  __shared__ float rs[8], rss[8];
  if ((tid & 31) == 0) { rs[tid >> 5] = s; rss[tid >> 5] = ss; }
  __syncthreads();
  float tot = 0.f, tots = 0.f;
#pragma unroll
  for (int i = 0; i < 8; ++i) { tot += rs[i]; tots += rss[i]; }
  const float mu  = tot * (1.0f / Dm);
  const float var = tots * (1.0f / Dm) - mu * mu;
  const float inv = rsqrtf(var + 1e-5f);
  const float* gr = g + tid * 4;
  uint2 pk;
  pk.x = (unsigned)f2bf((v.x - mu) * inv * gr[0]) | ((unsigned)f2bf((v.y - mu) * inv * gr[1]) << 16);
  pk.y = (unsigned)f2bf((v.z - mu) * inv * gr[2]) | ((unsigned)f2bf((v.w - mu) * inv * gr[3]) << 16);
  *reinterpret_cast<uint2*>(out + row * Dm + tid * 4) = pk;
}

// ---------------------------------------------------------------------------
// Main bf16 WMMA GEMM. Block tile 128(M) x 128(N), K-step 64, double-buffered
// TDM staging (one tensor_load_to_lds per tile, issued by wave 0).
// 8 waves as 2(M) x 4(N); wave tile 64x32 -> 16 WMMA per barrier.
//   accumulate=0: C = alpha*A@W + bias + res ; accumulate=1: C += alpha*A@W
//   outb (optional): bf16 copy of the final C value.
// ---------------------------------------------------------------------------
__global__ void __launch_bounds__(256)
gemm_bf16_kernel(const u16t* __restrict__ A, const u16t* __restrict__ W0,
                 long wStride, const int* __restrict__ subj,
                 const float* __restrict__ bias, const float* __restrict__ res,
                 float* __restrict__ C, u16t* __restrict__ outb,
                 float alpha, int accumulate, int K, int N) {
  __shared__ __align__(16) u16t As[2][128 * 64];  // 2 x 16KB
  __shared__ __align__(16) u16t Ws[2][64 * 128];  // 2 x 16KB
  const int b  = blockIdx.z;
  const int m0 = blockIdx.x * 128;
  const int n0 = blockIdx.y * 128;
  const u16t* W = W0 + (subj ? (long)subj[b] * wStride : 0);
  const long rowBase = (long)b * Tn + m0;
  const int tid  = threadIdx.x;
  const int lane = tid & 31, w = tid >> 5;
  const int l16  = lane & 15, half = lane >> 4;
  const int wm = w >> 2;  // 0..1 : 64-row sub-block
  const int wn = w & 3;   // 0..3 : 32-col sub-block

  auto stage = [&](int k0, int bufi) {
    // A tile: 128 rows x 64 K-elements, row stride K
    tdm_load_2d(As[bufi], A + rowBase * (long)K + k0, 64u, 128u,
                (unsigned long long)K);
    // W tile: 64 K-rows x 128 N-elements, row stride N
    tdm_load_2d(Ws[bufi], W + (long)k0 * N + n0, 128u, 64u,
                (unsigned long long)N);
  };

  v8f acc[4][2] = {};

  const int nk = K / 64;
  if (w == 0) stage(0, 0);
  for (int kk = 0; kk < nk; ++kk) {
    if (w == 0) wait_tensor();
    __syncthreads();
    if (w == 0 && kk + 1 < nk) stage((kk + 1) * 64, (kk + 1) & 1);
    const u16t* as = As[kk & 1];
    const u16t* ws = Ws[kk & 1];
#pragma unroll
    for (int ks = 0; ks < 2; ++ks) {  // two K=32 sub-steps
      FragB a[4], bb[2];
#pragma unroll
      for (int i = 0; i < 4; ++i) {
        const u16t* ar = as + (wm * 64 + i * 16 + l16) * 64 + ks * 32;
        a[i].u[0] = *reinterpret_cast<const uint4*>(ar + half * 8);
        a[i].u[1] = *reinterpret_cast<const uint4*>(ar + half * 8 + 16);
      }
#pragma unroll
      for (int j = 0; j < 2; ++j) {
        const u16t* wr = ws + (ks * 32 + half * 8) * 128 + wn * 32 + j * 16 + l16;
#pragma unroll
        for (int e = 0; e < 16; ++e)
          bb[j].v[e] = __builtin_bit_cast(__bf16, wr[((e & 7) + (e >> 3) * 16) * 128]);
      }
#pragma unroll
      for (int i = 0; i < 4; ++i)
#pragma unroll
        for (int j = 0; j < 2; ++j)
          acc[i][j] = wmma_bf16(a[i], bb[j], acc[i][j]);
    }
  }

#pragma unroll
  for (int i = 0; i < 4; ++i)
#pragma unroll
    for (int j = 0; j < 2; ++j)
#pragma unroll
      for (int jj = 0; jj < 8; ++jj) {
        long grow = rowBase + wm * 64 + i * 16 + half * 8 + jj;
        int  gcol = n0 + wn * 32 + j * 16 + l16;
        long idx  = grow * (long)N + gcol;
        float v = alpha * acc[i][j][jj];
        float r;
        if (accumulate) {
          r = C[idx] + v;
        } else {
          if (bias) v += bias[gcol];
          if (res)  v += res[idx];
          r = v;
        }
        C[idx] = r;
        if (outb) outb[idx] = f2bf(r);
      }
}

// ---------------------------------------------------------------------------
// Narrow GEMM for N == 64 (LoRA down-projection). Tile 128x64, 8 waves each
// owning 16 rows x 64 cols. Async single-buffered staging.
// ---------------------------------------------------------------------------
__global__ void __launch_bounds__(256)
gemm_n64_kernel(const u16t* __restrict__ A, const u16t* __restrict__ W0,
                long wStride, const int* __restrict__ subj,
                float* __restrict__ C, u16t* __restrict__ outb, int K) {
  const int N = 64;
  __shared__ __align__(16) u16t As[128 * 32];
  __shared__ __align__(16) u16t Ws[32 * 64];
  const int b  = blockIdx.z;
  const int m0 = blockIdx.x * 128;
  const u16t* W = W0 + (subj ? (long)subj[b] * wStride : 0);
  const long rowBase = (long)b * Tn + m0;
  const int tid  = threadIdx.x;
  const int lane = tid & 31, w = tid >> 5;
  const int l16  = lane & 15, half = lane >> 4;

  v8f acc[4] = {};

  for (int k0 = 0; k0 < K; k0 += 32) {
    __syncthreads();
#pragma unroll
    for (int it = 0; it < 2; ++it) {
      int c = tid + it * 256;  // 512 A chunks
      int r = c >> 2, cc = (c & 3) * 8;
      async_b128(As + r * 32 + cc, A + (rowBase + r) * (long)K + k0 + cc);
    }
    {
      int r = tid >> 3, cc = (tid & 7) * 8;  // 256 W chunks
      async_b128(Ws + r * 64 + cc, W + (long)(k0 + r) * N + cc);
    }
    wait_async();
    __syncthreads();
    FragB a;
    a.u[0] = *reinterpret_cast<const uint4*>(As + (w * 16 + l16) * 32 + half * 8);
    a.u[1] = *reinterpret_cast<const uint4*>(As + (w * 16 + l16) * 32 + half * 8 + 16);
#pragma unroll
    for (int nt = 0; nt < 4; ++nt) {
      FragB bf;
#pragma unroll
      for (int e = 0; e < 16; ++e)
        bf.v[e] = __builtin_bit_cast(__bf16, Ws[(half * 8 + (e & 7) + (e >> 3) * 16) * 64 + nt * 16 + l16]);
      acc[nt] = wmma_bf16(a, bf, acc[nt]);
    }
  }

#pragma unroll
  for (int nt = 0; nt < 4; ++nt)
#pragma unroll
    for (int j = 0; j < 8; ++j) {
      long grow = rowBase + w * 16 + half * 8 + j;
      int  gcol = nt * 16 + l16;
      long idx  = grow * (long)N + gcol;
      float v = acc[nt][j];
      C[idx] = v;
      if (outb) outb[idx] = f2bf(v);
    }
}

// ---------------------------------------------------------------------------
// Flash attention, bf16 WMMA, f32 online softmax.  64-key tiles.
//   qkv : bf16 [Bn, Tn, 3*Dm]; y : bf16 [Bn, Tn, Dm]
// Grid (Tn/128, Hn, Bn); 8 waves, each owning 16 query rows.
// ---------------------------------------------------------------------------
__global__ void __launch_bounds__(256)
attn_kernel(const u16t* __restrict__ qkv, u16t* __restrict__ y) {
  __shared__ __align__(16) u16t Kt[64 * 64];     // [dh][key] transposed (8KB)
  __shared__ __align__(16) u16t Vs[64 * 64];     // [key][dh] (8KB)
  __shared__ __align__(16) u16t Pw[8][16 * 64];  // per-wave P repack (16KB)

  const int b = blockIdx.z, h = blockIdx.y, q0 = blockIdx.x * 128;
  const int tid = threadIdx.x;
  const int lane = tid & 31, w = tid >> 5;
  const int l16 = lane & 15, half = lane >> 4;
  const long base = (long)b * Tn * 3072;
  const int hoff = h * 64;

  const int qrow = q0 + w * 16 + l16;
  const u16t* qp = qkv + base + (long)qrow * 3072 + hoff;
  FragB qa[2];
  qa[0].u[0] = *reinterpret_cast<const uint4*>(qp + half * 8);
  qa[0].u[1] = *reinterpret_cast<const uint4*>(qp + half * 8 + 16);
  qa[1].u[0] = *reinterpret_cast<const uint4*>(qp + 32 + half * 8);
  qa[1].u[1] = *reinterpret_cast<const uint4*>(qp + 32 + half * 8 + 16);

  v8f o[4] = {};
  float mrow[8], lrow[8];
#pragma unroll
  for (int j = 0; j < 8; ++j) { mrow[j] = -1e30f; lrow[j] = 0.f; }

  for (int t = 0; t < Tn; t += 64) {
    __syncthreads();
#pragma unroll
    for (int it = 0; it < 2; ++it) {  // V tile 64x64: async row-major
      int c = tid + it * 256;
      int key = c >> 3, dh = (c & 7) * 8;
      async_b128(Vs + key * 64 + dh,
                 qkv + base + (long)(t + key) * 3072 + 2048 + hoff + dh);
    }
    for (int i = tid; i < 4096; i += 256) {  // K tile: scalar transpose
      int key = i >> 6, dh = i & 63;
      Kt[dh * 64 + key] = qkv[base + (long)(t + key) * 3072 + 1024 + hoff + dh];
    }
    wait_async();
    __syncthreads();

    // S = Q * K^T : 4 column groups of 16 keys
    v8f sg[4] = {};
#pragma unroll
    for (int c = 0; c < 2; ++c) {
#pragma unroll
      for (int g = 0; g < 4; ++g) {
        FragB bk;
        const u16t* kr = Kt + (c * 32 + half * 8) * 64 + g * 16 + l16;
#pragma unroll
        for (int e = 0; e < 16; ++e)
          bk.v[e] = __builtin_bit_cast(__bf16, kr[((e & 7) + (e >> 3) * 16) * 64]);
        sg[g] = wmma_bf16(qa[c], bk, sg[g]);
      }
    }

    // online softmax over 64 keys
#pragma unroll
    for (int j = 0; j < 8; ++j) {
      float v = -1e30f;
#pragma unroll
      for (int g = 0; g < 4; ++g) { sg[g][j] *= 0.125f; v = fmaxf(v, sg[g][j]); }
      v = fmaxf(v, __shfl_xor(v, 1));
      v = fmaxf(v, __shfl_xor(v, 2));
      v = fmaxf(v, __shfl_xor(v, 4));
      v = fmaxf(v, __shfl_xor(v, 8));
      float mn  = fmaxf(mrow[j], v);
      float fac = __expf(mrow[j] - mn);
      mrow[j] = mn;
      float rsum = 0.f;
#pragma unroll
      for (int g = 0; g < 4; ++g) {
        float pv = __expf(sg[g][j] - mn);
        sg[g][j] = pv;
        rsum += pv;
      }
      rsum += __shfl_xor(rsum, 1);
      rsum += __shfl_xor(rsum, 2);
      rsum += __shfl_xor(rsum, 4);
      rsum += __shfl_xor(rsum, 8);
      lrow[j] = lrow[j] * fac + rsum;
      o[0][j] *= fac; o[1][j] *= fac; o[2][j] *= fac; o[3][j] *= fac;
    }

    // repack P (16x64) into A layout via per-wave LDS
    u16t* P = Pw[w];
#pragma unroll
    for (int j = 0; j < 8; ++j) {
      int row = half * 8 + j;
#pragma unroll
      for (int g = 0; g < 4; ++g) P[row * 64 + g * 16 + l16] = f2bf(sg[g][j]);
    }
    FragB pa[2];
#pragma unroll
    for (int ks = 0; ks < 2; ++ks) {
      pa[ks].u[0] = *reinterpret_cast<const uint4*>(P + l16 * 64 + ks * 32 + half * 8);
      pa[ks].u[1] = *reinterpret_cast<const uint4*>(P + l16 * 64 + ks * 32 + half * 8 + 16);
    }

    // O += P * V
#pragma unroll
    for (int nt = 0; nt < 4; ++nt) {
#pragma unroll
      for (int ks = 0; ks < 2; ++ks) {
        FragB bv;
        const u16t* vr = Vs + (ks * 32 + half * 8) * 64 + nt * 16 + l16;
#pragma unroll
        for (int e = 0; e < 16; ++e)
          bv.v[e] = __builtin_bit_cast(__bf16, vr[((e & 7) + (e >> 3) * 16) * 64]);
        o[nt] = wmma_bf16(pa[ks], bv, o[nt]);
      }
    }
  }

#pragma unroll
  for (int nt = 0; nt < 4; ++nt)
#pragma unroll
    for (int j = 0; j < 8; ++j) {
      long row = (long)b * Tn + q0 + w * 16 + half * 8 + j;
      int  col = hoff + nt * 16 + l16;
      y[row * Dm + col] = f2bf(o[nt][j] / lrow[j]);
    }
}

// ---------------------------------------------------------------------------
// Host orchestration
// ---------------------------------------------------------------------------
extern "C" void kernel_launch(void* const* d_in, const int* in_sizes, int n_in,
                              void* d_out, int out_size, void* d_ws, size_t ws_size,
                              hipStream_t stream) {
  const float* x     = (const float*)d_in[0];
  const int*   subj  = (const int*)d_in[1];
  const float* ln1w  = (const float*)d_in[3];
  const float* ln2w  = (const float*)d_in[4];
  const float* W_qkv = (const float*)d_in[5];
  const float* b_qkv = (const float*)d_in[6];
  const float* A_qkv = (const float*)d_in[7];
  const float* B_qkv = (const float*)d_in[8];
  const float* W_p   = (const float*)d_in[9];
  const float* b_p   = (const float*)d_in[10];
  const float* A_p   = (const float*)d_in[11];
  const float* B_p   = (const float*)d_in[12];
  const float* W_f1  = (const float*)d_in[13];
  const float* b_f1  = (const float*)d_in[14];
  const float* A_f1  = (const float*)d_in[15];
  const float* B_f1  = (const float*)d_in[16];
  const float* W_f2  = (const float*)d_in[17];
  const float* b_f2  = (const float*)d_in[18];
  const float* A_f2  = (const float*)d_in[19];
  const float* B_f2  = (const float*)d_in[20];
  float* out = (float*)d_out;

  char* p = (char*)d_ws;
  auto alloc = [&](size_t bytes) -> char* {
    char* r = p;
    p += (bytes + 255) & ~(size_t)255;
    return r;
  };

  u16t* wqkv_b = (u16t*)alloc((size_t)1024 * 3072 * 2);
  u16t* wp_b   = (u16t*)alloc((size_t)1024 * 1024 * 2);
  u16t* wf1_b  = (u16t*)alloc((size_t)1024 * 4096 * 2);
  u16t* wf2_b  = (u16t*)alloc((size_t)4096 * 1024 * 2);
  u16t* aqkv_b = (u16t*)alloc((size_t)10 * 1024 * 64 * 2);
  u16t* bqkv_b = (u16t*)alloc((size_t)10 * 64 * 3072 * 2);
  u16t* ap_b   = (u16t*)alloc((size_t)10 * 1024 * 64 * 2);
  u16t* bp_b   = (u16t*)alloc((size_t)10 * 64 * 1024 * 2);
  u16t* af1_b  = (u16t*)alloc((size_t)10 * 1024 * 64 * 2);
  u16t* bf1_b  = (u16t*)alloc((size_t)10 * 64 * 4096 * 2);
  u16t* af2_b  = (u16t*)alloc((size_t)10 * 4096 * 64 * 2);
  u16t* bf2_b  = (u16t*)alloc((size_t)10 * 64 * 1024 * 2);
  u16t*  xn_b  = (u16t*)alloc((size_t)BT * Dm * 2);
  float* qkv_f = (float*)alloc((size_t)BT * 3072 * 4);
  u16t*  qkv_b = (u16t*)alloc((size_t)BT * 3072 * 2);
  float* hl_f  = (float*)alloc((size_t)BT * Rk * 4);
  u16t*  hl_b  = (u16t*)alloc((size_t)BT * Rk * 2);
  u16t*  y_b   = (u16t*)alloc((size_t)BT * Dm * 2);
  float* x1_f  = (float*)alloc((size_t)BT * Dm * 4);
  float* h1_f  = (float*)alloc((size_t)BT * Dff * 4);
  u16t*  h1g_b = (u16t*)alloc((size_t)BT * Dff * 2);
  (void)ws_size; (void)in_sizes; (void)n_in; (void)out_size;

  auto cvt = [&](const float* in, u16t* o, long n) {
    cvt_kernel<<<2048, 256, 0, stream>>>(in, o, n);
  };
  auto gemm = [&](const u16t* A, const u16t* W, long wStride, const int* sj,
                  const float* bias, const float* res, float* C, u16t* outb,
                  float alpha, int accum, int K, int N) {
    dim3 g(Tn / 128, N / 128, Bn);
    gemm_bf16_kernel<<<g, 256, 0, stream>>>(A, W, wStride, sj, bias, res, C, outb,
                                            alpha, accum, K, N);
  };
  auto gemm64 = [&](const u16t* A, const u16t* W, long wStride, const int* sj,
                    float* C, u16t* outb, int K) {
    dim3 g(Tn / 128, 1, Bn);
    gemm_n64_kernel<<<g, 256, 0, stream>>>(A, W, wStride, sj, C, outb, K);
  };

  // weight conversion (per launch; no cross-call state)
  cvt(W_qkv, wqkv_b, 1024L * 3072);
  cvt(W_p,   wp_b,   1024L * 1024);
  cvt(W_f1,  wf1_b,  1024L * 4096);
  cvt(W_f2,  wf2_b,  4096L * 1024);
  cvt(A_qkv, aqkv_b, 10L * 1024 * 64);
  cvt(B_qkv, bqkv_b, 10L * 64 * 3072);
  cvt(A_p,   ap_b,   10L * 1024 * 64);
  cvt(B_p,   bp_b,   10L * 64 * 1024);
  cvt(A_f1,  af1_b,  10L * 1024 * 64);
  cvt(B_f1,  bf1_b,  10L * 64 * 4096);
  cvt(A_f2,  af2_b,  10L * 4096 * 64);
  cvt(B_f2,  bf2_b,  10L * 64 * 1024);

  // ---- attention block ----
  ln_kernel<<<BT, 256, 0, stream>>>(x, ln1w, xn_b);
  gemm(xn_b, wqkv_b, 0, nullptr, b_qkv, nullptr, qkv_f, nullptr, 1.f, 0, 1024, 3072);
  gemm64(xn_b, aqkv_b, 1024L * 64, subj, hl_f, hl_b, 1024);
  gemm(hl_b, bqkv_b, 64L * 3072, subj, nullptr, nullptr, qkv_f, qkv_b, LSCALE, 1, 64, 3072);
  attn_kernel<<<dim3(Tn / 128, Hn, Bn), 256, 0, stream>>>(qkv_b, y_b);
  gemm(y_b, wp_b, 0, nullptr, b_p, x, x1_f, nullptr, 1.f, 0, 1024, 1024);
  gemm64(y_b, ap_b, 1024L * 64, subj, hl_f, hl_b, 1024);
  gemm(hl_b, bp_b, 64L * 1024, subj, nullptr, nullptr, x1_f, nullptr, LSCALE, 1, 64, 1024);

  // ---- feed-forward block ----
  ln_kernel<<<BT, 256, 0, stream>>>(x1_f, ln2w, xn_b);
  gemm(xn_b, wf1_b, 0, nullptr, b_f1, nullptr, h1_f, nullptr, 1.f, 0, 1024, 4096);
  gemm64(xn_b, af1_b, 1024L * 64, subj, hl_f, hl_b, 1024);
  gemm(hl_b, bf1_b, 64L * 4096, subj, nullptr, nullptr, h1_f, nullptr, LSCALE, 1, 64, 4096);
  gelu_cvt_kernel<<<2048, 256, 0, stream>>>(h1_f, h1g_b, (long)BT * Dff);
  gemm(h1g_b, wf2_b, 0, nullptr, b_f2, x1_f, out, nullptr, 1.f, 0, 4096, 1024);
  gemm64(h1g_b, af2_b, 4096L * 64, subj, hl_f, hl_b, 4096);
  gemm(hl_b, bf2_b, 64L * 1024, subj, nullptr, nullptr, out, nullptr, LSCALE, 1, 64, 1024);
}
